// MoDLayer_25271587569795
// MI455X (gfx1250) — matrix operand order repaired and checked by reading, not compiled, not run on hardware
//
#include <hip/hip_runtime.h>
#include <math.h>

#define B_ 4
#define T_ 4096
#define HID_ 2048
#define HEADS_ 16
#define HD_ 128
#define INTER_ 8192
#define KSEL_ 512
#define BT_ (B_*T_)
#define ROWS_ (B_*KSEL_)   /* 2048 gathered tokens */

typedef __bf16 bf16;
typedef __attribute__((ext_vector_type(4)))  __bf16 v4bf;
typedef __attribute__((ext_vector_type(8)))  __bf16 v8bf;
typedef __attribute__((ext_vector_type(16))) __bf16 v16bf;
typedef __attribute__((ext_vector_type(8)))  float  v8f;

// ---------------------------------------------------------------------------
// Tiled WMMA GEMM: C[M,N] = epilogue(A[M,K] @ op(B)) with f32->bf16 LDS staging.
// TRANSB=false: Bm is N x K row-major (computes A @ W^T, the "x @ w.T" pattern)
// TRANSB=true : Bm is K x N row-major (computes A @ B, used for attn @ V)
// Batched via blockIdx.z with (outer,inner) strides: base = (z/zdiv)*So + (z%zdiv)*Si
// LDS tiles are stored K-contiguous per row with a 40-element (80 B) row stride:
// 16B-aligned for ds_load_b128 fragment reads, conflict-spread across 64 banks.
// ---------------------------------------------------------------------------
enum { EP_BIAS = 0, EP_GELU = 1, EP_SILU_MUL = 2, EP_SCALE_MASK = 3, EP_RES = 4 };

#define LDSTR 40

template<int EP, bool TRANSB>
__global__ __launch_bounds__(256)
void gemm_wmma(const float* __restrict__ A,  long long aSo, long long aSi,
               const float* __restrict__ Bm, long long bSo, long long bSi,
               float* __restrict__ C,        long long cSo, long long cSi,
               const float* __restrict__ bias,
               const float* __restrict__ extra, long long eSo, long long eSi,
               int zdiv, int M, int N, int K,
               int lda, int ldb, int ldc, float alpha)
{
  __shared__ bf16 As[128 * LDSTR];   // [m][k]
  __shared__ bf16 Bs[128 * LDSTR];   // [n][k]

  const int z  = blockIdx.z;
  const long long zo = z / zdiv, zi = z % zdiv;
  A  += zo * aSo + zi * aSi;
  Bm += zo * bSo + zi * bSi;
  C  += zo * cSo + zi * cSi;
  const float* Ex = extra ? (extra + zo * eSo + zi * eSi) : nullptr;

  const int tid  = threadIdx.x;
  const int lane = tid & 31;
  const int wave = tid >> 5;           // 8 waves
  const int wm   = (wave & 3) * 32;    // 4 waves along M, 32 rows each
  const int wn   = (wave >> 2) * 64;   // 2 waves along N, 64 cols each
  const int mt   = blockIdx.y * 128;
  const int nt   = blockIdx.x * 128;

  v8f acc[2][4];
  #pragma unroll
  for (int i = 0; i < 2; ++i)
    #pragma unroll
    for (int j = 0; j < 4; ++j) { v8f zv = {}; acc[i][j] = zv; }

  float4 pa[4], pb[4];   // register-staged next tile (prefetch pipeline)

  auto fetchA = [&](int k0) {
    #pragma unroll
    for (int c = 0; c < 4; ++c) {
      int lin = tid + c * 256;                       // 0..1023
      int m = lin >> 3, kc = (lin & 7) << 2;         // 128 rows x 8 float4
      pa[c] = *(const float4*)(A + (long long)(mt + m) * lda + (k0 + kc));
    }
  };
  auto fetchB = [&](int k0) {
    #pragma unroll
    for (int c = 0; c < 4; ++c) {
      int lin = tid + c * 256;
      if (TRANSB) {
        int kk = lin >> 5, nc = (lin & 31) << 2;     // 32 k-rows x 32 float4
        pb[c] = *(const float4*)(Bm + (long long)(k0 + kk) * ldb + (nt + nc));
      } else {
        int n = lin >> 3, kc = (lin & 7) << 2;       // 128 n-rows x 8 float4
        pb[c] = *(const float4*)(Bm + (long long)(nt + n) * ldb + (k0 + kc));
      }
    }
  };
  auto stage = [&]() {
    #pragma unroll
    for (int c = 0; c < 4; ++c) {
      int lin = tid + c * 256;
      {
        int m = lin >> 3, kc = (lin & 7) << 2;
        v4bf t = { (bf16)pa[c].x, (bf16)pa[c].y, (bf16)pa[c].z, (bf16)pa[c].w };
        *(v4bf*)&As[m * LDSTR + kc] = t;
      }
      if (TRANSB) {
        int kk = lin >> 5, nc = (lin & 31) << 2;
        Bs[(nc + 0) * LDSTR + kk] = (bf16)pb[c].x;
        Bs[(nc + 1) * LDSTR + kk] = (bf16)pb[c].y;
        Bs[(nc + 2) * LDSTR + kk] = (bf16)pb[c].z;
        Bs[(nc + 3) * LDSTR + kk] = (bf16)pb[c].w;
      } else {
        int n = lin >> 3, kc = (lin & 7) << 2;
        v4bf t = { (bf16)pb[c].x, (bf16)pb[c].y, (bf16)pb[c].z, (bf16)pb[c].w };
        *(v4bf*)&Bs[n * LDSTR + kc] = t;
      }
    }
  };

  fetchA(0); fetchB(0);

  for (int k0 = 0; k0 < K; k0 += 32) {
    stage();
    __syncthreads();
    if (k0 + 32 < K) { fetchA(k0 + 32); fetchB(k0 + 32); }  // overlap with WMMA

    // Fragments per CDNA5 16-bit layouts (ISA 7.12.2), vector LDS reads:
    // A 16x32: lanes 0-15 row M=lane, K {0..7,16..23}; lanes 16-31 K {8..15,24..31}
    v16bf af[2], bfv[4];
    const int ar  = lane & 15;
    const int kbA = (lane >> 4) * 8;
    #pragma unroll
    for (int mi = 0; mi < 2; ++mi) {
      int r = wm + mi * 16 + ar;
      v8bf lo = *(const v8bf*)&As[r * LDSTR + kbA];
      v8bf hi = *(const v8bf*)&As[r * LDSTR + kbA + 16];
      af[mi] = __builtin_shufflevector(lo, hi, 0,1,2,3,4,5,6,7,8,9,10,11,12,13,14,15);
    }
    // B 32x16: lanes 0-15 col N=lane, K {0..7,16..23}; lanes 16-31 K {8..15,24..31}
    const int bc  = lane & 15;
    const int kbB = (lane >> 4) * 8;
    #pragma unroll
    for (int ni = 0; ni < 4; ++ni) {
      int cix = wn + ni * 16 + bc;
      v8bf lo = *(const v8bf*)&Bs[cix * LDSTR + kbB];
      v8bf hi = *(const v8bf*)&Bs[cix * LDSTR + kbB + 16];
      bfv[ni] = __builtin_shufflevector(lo, hi, 0,1,2,3,4,5,6,7,8,9,10,11,12,13,14,15);
    }

    #pragma unroll
    for (int mi = 0; mi < 2; ++mi)
      #pragma unroll
      for (int ni = 0; ni < 4; ++ni)
        acc[mi][ni] = __builtin_amdgcn_wmma_f32_16x16x32_bf16(
            false, af[mi], false, bfv[ni], (short)0, acc[mi][ni], false, false);
    __syncthreads();
  }

  // Store with epilogue. C/D layout: VGPR r -> row r + 8*(lane>=16), col = lane&15
  const int rhi = (lane >> 4) * 8;
  const int cl  = lane & 15;
  #pragma unroll
  for (int mi = 0; mi < 2; ++mi) {
    #pragma unroll
    for (int ni = 0; ni < 4; ++ni) {
      #pragma unroll
      for (int r = 0; r < 8; ++r) {
        int row = mt + wm + mi * 16 + r + rhi;
        int col = nt + wn + ni * 16 + cl;
        float v = acc[mi][ni][r];
        if (EP == EP_BIAS) {
          if (bias) v += bias[col];
        } else if (EP == EP_GELU) {
          v += bias[col];
          v = 0.5f * v * (1.0f + erff(v * 0.70710678118f));
        } else if (EP == EP_SILU_MUL) {
          v = v / (1.0f + __expf(-v));
          v *= Ex[(long long)row * ldc + col];
        } else if (EP == EP_SCALE_MASK) {
          v *= alpha;
          if (col > row) v = -1.0e9f;
        } else if (EP == EP_RES) {
          v += Ex[(long long)row * ldc + col];
        }
        C[(long long)row * ldc + col] = v;
      }
    }
  }
}

// ---------------------------------------------------------------------------
// Small helper kernels
// ---------------------------------------------------------------------------
__global__ void rowdot_kernel(const float* __restrict__ X, const float* __restrict__ w,
                              const float* __restrict__ bscalar, float* __restrict__ out, int K)
{
  __shared__ float red[256];
  const float* x = X + (long long)blockIdx.x * K;
  float s = 0.f;
  for (int i = threadIdx.x; i < K; i += 256) s += x[i] * w[i];
  red[threadIdx.x] = s; __syncthreads();
  for (int o = 128; o; o >>= 1) { if (threadIdx.x < o) red[threadIdx.x] += red[threadIdx.x + o]; __syncthreads(); }
  if (!threadIdx.x) out[blockIdx.x] = red[0] + (bscalar ? bscalar[0] : 0.0f);
}

__global__ void fill_kernel(float* __restrict__ p, float v, long long n)
{
  for (long long i = blockIdx.x * 256ll + threadIdx.x; i < n; i += (long long)gridDim.x * 256)
    p[i] = v;
}

// Per-batch top-512-of-4096: 8-pass 4-bit radix select on order-preserving uint
// keys, tie fill, then bitonic sort of the selected indices (ascending).
__global__ __launch_bounds__(1024)
void topk_kernel(const float* __restrict__ logits, int* __restrict__ idx_out,
                 float* __restrict__ gate_out, float* __restrict__ mask)
{
  const int b = blockIdx.x;
  const float* lg = logits + (long long)b * T_;
  __shared__ int hist[16];
  __shared__ unsigned sh_prefix;
  __shared__ int sh_need, sh_sel, sh_eq;
  __shared__ int sidx[KSEL_];
  const int tid = threadIdx.x;
  if (tid == 0) { sh_prefix = 0u; sh_need = KSEL_; }
  __syncthreads();

  for (int p = 7; p >= 0; --p) {
    if (tid < 16) hist[tid] = 0;
    __syncthreads();
    unsigned pref = sh_prefix;
    int shift = p * 4;
    unsigned hiMask = (p == 7) ? 0u : (0xFFFFFFFFu << (shift + 4));
    for (int t = tid; t < T_; t += 1024) {
      unsigned u = __float_as_uint(lg[t]);
      unsigned key = (u & 0x80000000u) ? ~u : (u | 0x80000000u);
      if ((key & hiMask) == (pref & hiMask))
        atomicAdd(&hist[(key >> shift) & 15], 1);
    }
    __syncthreads();
    if (tid == 0) {
      int need = sh_need, cum = 0, d = 15;
      for (; d >= 0; --d) { if (cum + hist[d] >= need) break; cum += hist[d]; }
      if (d < 0) d = 0;
      sh_prefix = pref | ((unsigned)d << shift);
      sh_need = need - cum;
    }
    __syncthreads();
  }
  const unsigned thr = sh_prefix;
  const int needEq = sh_need;
  if (tid == 0) { sh_sel = 0; sh_eq = 0; }
  __syncthreads();
  for (int t = tid; t < T_; t += 1024) {
    unsigned u = __float_as_uint(lg[t]);
    unsigned key = (u & 0x80000000u) ? ~u : (u | 0x80000000u);
    if (key > thr) {
      int p2 = atomicAdd(&sh_sel, 1);
      if (p2 < KSEL_) sidx[p2] = t;
    } else if (key == thr) {
      int e = atomicAdd(&sh_eq, 1);
      if (e < needEq) { int p2 = atomicAdd(&sh_sel, 1); if (p2 < KSEL_) sidx[p2] = t; }
    }
  }
  __syncthreads();
  // bitonic sort (512 ints ascending)
  for (int kk = 2; kk <= KSEL_; kk <<= 1) {
    for (int j = kk >> 1; j > 0; j >>= 1) {
      int i = tid;
      if (i < KSEL_) {
        int ixj = i ^ j;
        if (ixj > i) {
          bool up = ((i & kk) == 0);
          int a = sidx[i], c = sidx[ixj];
          if ((a > c) == up) { sidx[i] = c; sidx[ixj] = a; }
        }
      }
      __syncthreads();
    }
  }
  for (int i = tid; i < KSEL_; i += 1024) {
    int t = sidx[i];
    idx_out[b * KSEL_ + i] = t;
    float l = lg[t];
    gate_out[b * KSEL_ + i] = 1.0f / (1.0f + __expf(-l));
    mask[(long long)b * T_ + t] = 1.0f;
  }
}

__global__ void bce_kernel(const float* __restrict__ lg, const float* __restrict__ tgt,
                           float* __restrict__ out, float invN, long long n)
{
  __shared__ float red[256];
  float s = 0.f;
  for (long long i = blockIdx.x * 256ll + threadIdx.x; i < n; i += (long long)gridDim.x * 256) {
    float l = lg[i], y = tgt[i];
    s += fmaxf(l, 0.f) - l * y + log1pf(__expf(-fabsf(l)));
  }
  red[threadIdx.x] = s; __syncthreads();
  for (int o = 128; o; o >>= 1) { if (threadIdx.x < o) red[threadIdx.x] += red[threadIdx.x + o]; __syncthreads(); }
  if (!threadIdx.x) atomicAdd(out, red[0] * invN);
}

// RMSNorm over HID_ cols; optional gather (idx) and raw-row copy (xcopy).
__global__ void rmsnorm_kernel(const float* __restrict__ in, const int* __restrict__ idx,
                               const float* __restrict__ w, float* __restrict__ xcopy,
                               float* __restrict__ out)
{
  __shared__ float red[256];
  const int row = blockIdx.x;
  const float* src;
  if (idx) { int b = row / KSEL_; src = in + ((long long)b * T_ + idx[row]) * HID_; }
  else     { src = in + (long long)row * HID_; }
  float s = 0.f;
  for (int i = threadIdx.x; i < HID_; i += 256) { float v = src[i]; s += v * v; }
  red[threadIdx.x] = s; __syncthreads();
  for (int o = 128; o; o >>= 1) { if (threadIdx.x < o) red[threadIdx.x] += red[threadIdx.x + o]; __syncthreads(); }
  float r = rsqrtf(red[0] / (float)HID_ + 1e-6f);
  for (int i = threadIdx.x; i < HID_; i += 256) {
    float v = src[i];
    if (xcopy) xcopy[(long long)row * HID_ + i] = v;
    out[(long long)row * HID_ + i] = v * r * w[i];
  }
}

__global__ void rope_kernel(float* __restrict__ q, float* __restrict__ k, const int* __restrict__ idx)
{
  const int row = blockIdx.x;
  const float pos = (float)idx[row];
  const long long base = (long long)row * HID_;
  for (int i = threadIdx.x; i < HEADS_ * 64; i += 256) {
    int h = i >> 6, d = i & 63;
    float inv = __powf(10000.0f, -(float)d / 64.0f);
    float ang = pos * inv;
    float c = __cosf(ang), s = __sinf(ang);
    long long o1 = base + h * HD_ + d, o2 = o1 + 64;
    float q1 = q[o1], q2 = q[o2];
    q[o1] = q1 * c - q2 * s; q[o2] = q2 * c + q1 * s;
    float k1 = k[o1], k2 = k[o2];
    k[o1] = k1 * c - k2 * s; k[o2] = k2 * c + k1 * s;
  }
}

__global__ void softmax_kernel(float* __restrict__ sc)
{
  __shared__ float red[256];
  float* p = sc + (long long)blockIdx.x * KSEL_;
  float m = -1e30f;
  for (int i = threadIdx.x; i < KSEL_; i += 256) m = fmaxf(m, p[i]);
  red[threadIdx.x] = m; __syncthreads();
  for (int o = 128; o; o >>= 1) { if (threadIdx.x < o) red[threadIdx.x] = fmaxf(red[threadIdx.x], red[threadIdx.x + o]); __syncthreads(); }
  m = red[0]; __syncthreads();
  float s = 0.f;
  for (int i = threadIdx.x; i < KSEL_; i += 256) { float e = __expf(p[i] - m); p[i] = e; s += e; }
  red[threadIdx.x] = s; __syncthreads();
  for (int o = 128; o; o >>= 1) { if (threadIdx.x < o) red[threadIdx.x] += red[threadIdx.x + o]; __syncthreads(); }
  float inv = 1.0f / red[0];
  for (int i = threadIdx.x; i < KSEL_; i += 256) p[i] *= inv;
}

__global__ void copy4_kernel(const float4* __restrict__ src, float4* __restrict__ dst, long long n4)
{
  for (long long i = blockIdx.x * 256ll + threadIdx.x; i < n4; i += (long long)gridDim.x * 256)
    dst[i] = src[i];
}

__global__ void scatter_kernel(const float* __restrict__ x, const float* __restrict__ lo,
                               const float* __restrict__ gate, const int* __restrict__ idx,
                               float* __restrict__ out)
{
  const int row = blockIdx.x;
  const int b = row / KSEL_;
  const float g = gate[row];
  const long long dst = ((long long)b * T_ + idx[row]) * HID_;
  const long long src = (long long)row * HID_;
  for (int i = threadIdx.x; i < HID_; i += 256) {
    float xv = x[src + i];
    out[dst + i] = xv + g * (lo[src + i] - xv);
  }
}

// ---------------------------------------------------------------------------
extern "C" void kernel_launch(void* const* d_in, const int* in_sizes, int n_in,
                              void* d_out, int out_size, void* d_ws, size_t ws_size,
                              hipStream_t stream)
{
  (void)in_sizes; (void)n_in; (void)out_size; (void)ws_size;
  const float* hidden  = (const float*)d_in[0];
  const float* routerw = (const float*)d_in[1];
  const float* cfc1w   = (const float*)d_in[2];
  const float* cfc1b   = (const float*)d_in[3];
  const float* cfc2w   = (const float*)d_in[4];
  const float* cfc2b   = (const float*)d_in[5];
  const float* ln1w    = (const float*)d_in[6];
  const float* ln2w    = (const float*)d_in[7];
  const float* qw = (const float*)d_in[8];  const float* qb = (const float*)d_in[9];
  const float* kw = (const float*)d_in[10]; const float* kb = (const float*)d_in[11];
  const float* vw = (const float*)d_in[12]; const float* vb = (const float*)d_in[13];
  const float* ow = (const float*)d_in[14];
  const float* gatew = (const float*)d_in[15];
  const float* upw   = (const float*)d_in[16];
  const float* downw = (const float*)d_in[17];

  float* out = (float*)d_out;
  const long long NS = (long long)BT_ * HID_;   // new_states elements

  // workspace bump allocation (floats)
  float* ws = (float*)d_ws;
  size_t off = 0;
  auto alloc = [&](size_t n) { float* p = ws + off; off += n; return p; };
  float* logits = alloc(BT_);
  float* predl  = alloc(BT_);
  float* mask   = alloc(BT_);
  float* gate   = alloc(ROWS_);
  int*   idx    = (int*)alloc(ROWS_);
  float* region = alloc((size_t)16 * 1024 * 1024);       // predictor hidden, then attn scores
  float* xbuf   = alloc((size_t)ROWS_ * HID_);
  float* hbuf   = alloc((size_t)ROWS_ * HID_);
  float* qbuf   = alloc((size_t)ROWS_ * HID_);
  float* kbuf   = alloc((size_t)ROWS_ * HID_);
  float* vbuf   = alloc((size_t)ROWS_ * HID_);
  float* aobuf  = alloc((size_t)ROWS_ * HID_);
  float* h1buf  = alloc((size_t)ROWS_ * HID_);
  float* h2buf  = alloc((size_t)ROWS_ * HID_);           // later reused as layer_out
  float* gbuf   = alloc((size_t)ROWS_ * INTER_);

  const float rsH = 0.08838834764831845f;  // 1/sqrt(128)

  // 1) router logits + top-k + mask
  rowdot_kernel<<<BT_, 256, 0, stream>>>(hidden, routerw, nullptr, logits, HID_);
  fill_kernel<<<64, 256, 0, stream>>>(mask, 0.0f, BT_);
  topk_kernel<<<B_, 1024, 0, stream>>>(logits, idx, gate, mask);

  // 2) predictor: GELU(hidden @ cfc1^T + b) then dot cfc2
  gemm_wmma<EP_GELU, false><<<dim3(512 / 128, BT_ / 128, 1), 256, 0, stream>>>(
      hidden, 0, 0, cfc1w, 0, 0, region, 0, 0, cfc1b, nullptr, 0, 0,
      1, BT_, 512, HID_, HID_, HID_, 512, 0.f);
  rowdot_kernel<<<BT_, 256, 0, stream>>>(region, cfc2w, cfc2b, predl, 512);

  // 3) losses into d_out tail
  fill_kernel<<<1, 32, 0, stream>>>(out + NS, 0.0f, 2);
  bce_kernel<<<256, 256, 0, stream>>>(logits, mask, out + NS,     1.0f / (float)BT_, BT_);
  bce_kernel<<<256, 256, 0, stream>>>(predl,  mask, out + NS + 1, 1.0f / (float)BT_, BT_);

  // 4) gather + RMSNorm1
  rmsnorm_kernel<<<ROWS_, 256, 0, stream>>>(hidden, idx, ln1w, xbuf, hbuf);

  // 5) QKV projections
  gemm_wmma<EP_BIAS, false><<<dim3(16, 16, 1), 256, 0, stream>>>(
      hbuf, 0, 0, qw, 0, 0, qbuf, 0, 0, qb, nullptr, 0, 0,
      1, ROWS_, HID_, HID_, HID_, HID_, HID_, 0.f);
  gemm_wmma<EP_BIAS, false><<<dim3(16, 16, 1), 256, 0, stream>>>(
      hbuf, 0, 0, kw, 0, 0, kbuf, 0, 0, kb, nullptr, 0, 0,
      1, ROWS_, HID_, HID_, HID_, HID_, HID_, 0.f);
  gemm_wmma<EP_BIAS, false><<<dim3(16, 16, 1), 256, 0, stream>>>(
      hbuf, 0, 0, vw, 0, 0, vbuf, 0, 0, vb, nullptr, 0, 0,
      1, ROWS_, HID_, HID_, HID_, HID_, HID_, 0.f);
  rope_kernel<<<ROWS_, 256, 0, stream>>>(qbuf, kbuf, idx);

  // 6) attention scores: per (b,h) GEMM 512x512x128 with scale + causal mask
  const long long perB = (long long)KSEL_ * HID_;        // 1048576
  const long long perZ = (long long)KSEL_ * KSEL_;       // 262144
  gemm_wmma<EP_SCALE_MASK, false><<<dim3(4, 4, B_ * HEADS_), 256, 0, stream>>>(
      qbuf, perB, 128, kbuf, perB, 128, region, (long long)HEADS_ * perZ, perZ,
      nullptr, nullptr, 0, 0,
      HEADS_, KSEL_, KSEL_, HD_, HID_, HID_, KSEL_, rsH);
  softmax_kernel<<<B_ * HEADS_ * KSEL_, 256, 0, stream>>>(region);

  // 7) attn @ V (TRANSB: B is KxN with ld=HID_)
  gemm_wmma<EP_BIAS, true><<<dim3(1, 4, B_ * HEADS_), 256, 0, stream>>>(
      region, (long long)HEADS_ * perZ, perZ, vbuf, perB, 128, aobuf, perB, 128,
      nullptr, nullptr, 0, 0,
      HEADS_, KSEL_, HD_, KSEL_, KSEL_, HID_, HID_, 0.f);

  // 8) O projection + residual (h1 = x + ao @ o^T)
  gemm_wmma<EP_RES, false><<<dim3(16, 16, 1), 256, 0, stream>>>(
      aobuf, 0, 0, ow, 0, 0, h1buf, 0, 0, nullptr, xbuf, 0, 0,
      1, ROWS_, HID_, HID_, HID_, HID_, HID_, 0.f);

  // 9) RMSNorm2 + MLP: gbuf = silu(h2@gate^T) * (h2@up^T); layer_out = h1 + gbuf@down^T
  rmsnorm_kernel<<<ROWS_, 256, 0, stream>>>(h1buf, nullptr, ln2w, nullptr, h2buf);
  gemm_wmma<EP_BIAS, false><<<dim3(64, 16, 1), 256, 0, stream>>>(
      h2buf, 0, 0, upw, 0, 0, gbuf, 0, 0, nullptr, nullptr, 0, 0,
      1, ROWS_, INTER_, HID_, HID_, HID_, INTER_, 0.f);
  gemm_wmma<EP_SILU_MUL, false><<<dim3(64, 16, 1), 256, 0, stream>>>(
      h2buf, 0, 0, gatew, 0, 0, gbuf, 0, 0, nullptr, gbuf, 0, 0,
      1, ROWS_, INTER_, HID_, HID_, HID_, INTER_, 0.f);
  gemm_wmma<EP_RES, false><<<dim3(16, 16, 1), 256, 0, stream>>>(
      gbuf, 0, 0, downw, 0, 0, h2buf, 0, 0, nullptr, h1buf, 0, 0,
      1, ROWS_, HID_, INTER_, INTER_, INTER_, HID_, 0.f);

  // 10) new_states = hidden; scatter gated update rows
  copy4_kernel<<<4096, 256, 0, stream>>>((const float4*)hidden, (float4*)out, NS / 4);
  scatter_kernel<<<ROWS_, 256, 0, stream>>>(xbuf, h2buf, gate, idx, out);
}
